// AttentionWindow_39152921870888
// MI455X (gfx1250) — matrix operand order repaired and checked by reading, not compiled, not run on hardware
//
#include <hip/hip_runtime.h>
#include <hip/hip_bf16.h>

typedef __bf16 bf16_t;
typedef __attribute__((ext_vector_type(16))) __bf16 bf16x16;
typedef __attribute__((ext_vector_type(8)))  __bf16 bf16x8;
typedef __attribute__((ext_vector_type(8)))  float  f32x8;

#define NEG_MAX (-3.402823466e38f)

// ---------------------------------------------------------------------------
// WMMA helpers (wave32, 16x16x32 bf16, fp32 accumulate)
// ---------------------------------------------------------------------------
__device__ __forceinline__ f32x8 wmma_bf16(bf16x16 a, bf16x16 b, f32x8 c) {
    return __builtin_amdgcn_wmma_f32_16x16x32_bf16(
        /*neg_a=*/false, a, /*neg_b=*/false, b,
        /*c_mod=*/(short)0, c, /*reuse_a=*/false, /*reuse_b=*/false);
}

// A-matrix 16x32 bf16 fragment.
// ISA layout: lane<16 -> row M=lane, K = kbase+{0..7} and kbase+{16..23}
//             lane>=16 -> row M=lane-16, K = kbase+{8..15} and kbase+{24..31}
// source: row-major [M][K], stride in elements (rows 16B aligned)
__device__ __forceinline__ bf16x16 load_a_frag(const bf16_t* base, int stride,
                                               int lane, int kbase) {
    const int m   = lane & 15;
    const int sel = (lane >> 4) * 8;
    const bf16_t* p = base + m * stride + kbase + sel;
    bf16x8 lo = *(const bf16x8*)p;
    bf16x8 hi = *(const bf16x8*)(p + 16);
    bf16x16 r;
#pragma unroll
    for (int i = 0; i < 8; ++i) { r[i] = lo[i]; r[8 + i] = hi[i]; }
    return r;
}

// B-matrix 32x16 bf16 fragment.
// ISA layout: lane<16 -> column N=lane, K = kbase+{0..7} and kbase+{16..23}
//             lane>=16 -> column N=lane-16, K = kbase+{8..15} and kbase+{24..31}
// source: column n contiguous in K at base + n*stride (i.e. K-major per column)
__device__ __forceinline__ bf16x16 load_b_frag(const bf16_t* base, int stride,
                                               int lane, int kbase) {
    const int n     = lane & 15;
    const int khalf = (lane >> 4) * 8;
    const bf16_t* p = base + n * stride + kbase + khalf;
    bf16x8 lo = *(const bf16x8*)p;
    bf16x8 hi = *(const bf16x8*)(p + 16);
    bf16x16 r;
#pragma unroll
    for (int i = 0; i < 8; ++i) { r[i] = lo[i]; r[8 + i] = hi[i]; }
    return r;
}

__device__ __forceinline__ f32x8 zero8() {
    f32x8 z = {0.f, 0.f, 0.f, 0.f, 0.f, 0.f, 0.f, 0.f};
    return z;
}

// ---------------------------------------------------------------------------
// Pre-pass: fp32 [K][N] -> bf16 [N][K] (transpose+convert), coalesced writes
// ---------------------------------------------------------------------------
__global__ void transpose_to_bf16_kernel(const float* __restrict__ in,
                                         bf16_t* __restrict__ out,
                                         int K, int N) {
    int idx = blockIdx.x * blockDim.x + threadIdx.x;
    if (idx >= K * N) return;
    int n = idx / K;
    int k = idx % K;
    out[idx] = (bf16_t)in[k * N + n];
}

// ---------------------------------------------------------------------------
// Fused windowed attention: one block per window (1024 blocks, 8 waves)
// ---------------------------------------------------------------------------
// LDS layout (bytes):
//   Xb/Ob : [80][264] bf16  @ 0        (42240)   X tile, reused for O
//   Qb    : [80][264] bf16  @ 42240    (42240)   pre-scaled by 1/8
//   Kb    : [80][264] bf16  @ 84480    (42240)
//   Vt    : [256][96] bf16  @ 126720   (49152)   V transposed: [head*64+d][token]
//   Pf    : [80][96] f32    @ 175872   (30720)   scores (bias+mask applied)
//   Pb    : [80][96] bf16   @ 206592   (15360)   softmaxed probs, K padded to 96
#define SMEM_BYTES 221952
#define XS 264   // X/Q/K/O row stride (bf16 elements)
#define PS 96    // P / Vt row stride

__global__ __launch_bounds__(256) void attn_win_kernel(
    const float* __restrict__ x,        // (4, 256, 128, 128)
    const bf16_t* __restrict__ wqkvT,   // (768, 256) bf16
    const bf16_t* __restrict__ woutT,   // (256, 256) bf16
    const float* __restrict__ bias_table, // (441, 4)
    float* __restrict__ out) {          // (4, 256, 128, 128)

    extern __shared__ char smem[];
    bf16_t* Xb = (bf16_t*)(smem);
    bf16_t* Qb = (bf16_t*)(smem + 42240);
    bf16_t* Kb = (bf16_t*)(smem + 84480);
    bf16_t* Vt = (bf16_t*)(smem + 126720);
    float*  Pf = (float*)(smem + 175872);
    bf16_t* Pb = (bf16_t*)(smem + 206592);
    bf16_t* Ob = Xb;  // reuse after QKV stage

    const int blk  = blockIdx.x;
    const int wh   = blk >> 5;     // window row 0..31
    const int ww   = blk & 31;     // window col 0..31
    const int tid  = threadIdx.x;
    const int lane = tid & 31;
    const int wv   = tid >> 5;     // wave id 0..7

    // ---- zero padded regions -------------------------------------------
    // X rows 64..79 (padded agent 4) incl. pad columns
    for (int i = tid; i < 16 * XS; i += 256) Xb[64 * XS + i] = (bf16_t)0.0f;
    // Vt token tail 80..95 for every (head,d)
    for (int i = tid; i < 256 * 16; i += 256) {
        int col = i >> 4;
        int j   = i & 15;
        Vt[col * PS + 80 + j] = (bf16_t)0.0f;
    }

    // ---- gather x -> X tile (bf16) -------------------------------------
    // 4096 float4 loads: idx -> (agent a, row ph, channel c), 4 pw values
    for (int it = 0; it < 16; ++it) {
        int idx = it * 256 + tid;          // 0..4095
        int a   = idx >> 10;               // 0..3
        int rem = idx & 1023;
        int ph  = rem >> 8;                // 0..3
        int c   = rem & 255;
        const float4 src =
            *(const float4*)(x + (((a * 256 + c) * 128) + (wh * 4 + ph)) * 128 + ww * 4);
        int t0 = a * 16 + ph * 4;
        Xb[(t0 + 0) * XS + c] = (bf16_t)src.x;
        Xb[(t0 + 1) * XS + c] = (bf16_t)src.y;
        Xb[(t0 + 2) * XS + c] = (bf16_t)src.z;
        Xb[(t0 + 3) * XS + c] = (bf16_t)src.w;
    }
    __syncthreads();

    // ---- QKV projection: [80x256] @ [256x768] --------------------------
    // 48 n-tiles split across 8 waves (6 each); 5 m-tiles; K = 8 steps of 32
    for (int j = 0; j < 6; ++j) {
        const int nt = wv + 8 * j;
        const int n0 = nt * 16;
        f32x8 acc[5];
#pragma unroll
        for (int mt = 0; mt < 5; ++mt) acc[mt] = zero8();
        for (int ks = 0; ks < 8; ++ks) {
            bf16x16 bfr = load_b_frag(wqkvT + n0 * 256, 256, lane, ks * 32);
#pragma unroll
            for (int mt = 0; mt < 5; ++mt) {
                bf16x16 afr = load_a_frag(Xb + mt * 16 * XS, XS, lane, ks * 32);
                acc[mt] = wmma_bf16(afr, bfr, acc[mt]);
            }
        }
        const int n   = n0 + (lane & 15);
        const int sel = (lane >> 4) * 8;
#pragma unroll
        for (int mt = 0; mt < 5; ++mt) {
#pragma unroll
            for (int r = 0; r < 8; ++r) {
                int m = mt * 16 + sel + r;
                float v = acc[mt][r];
                if (n < 256) {
                    Qb[m * XS + n] = (bf16_t)(v * 0.125f);      // scale = dh^-0.5
                } else if (n < 512) {
                    Kb[m * XS + (n - 256)] = (bf16_t)v;
                } else {
                    Vt[(n - 512) * PS + m] = (bf16_t)v;          // transposed V
                }
            }
        }
    }
    __syncthreads();

    // ---- attention per head --------------------------------------------
    for (int h = 0; h < 4; ++h) {
        // S = Q Kt : 5x5 = 25 tiles of 16x16, K = 64 (2 steps)
        for (int tt = wv; tt < 25; tt += 8) {
            const int mt = tt / 5, nt = tt % 5;
            f32x8 acc = zero8();
            for (int ks = 0; ks < 2; ++ks) {
                bf16x16 afr = load_a_frag(Qb + mt * 16 * XS + h * 64, XS, lane, ks * 32);
                bf16x16 bfr = load_b_frag(Kb + nt * 16 * XS + h * 64, XS, lane, ks * 32);
                acc = wmma_bf16(afr, bfr, acc);
            }
            // relative-position bias + agent padding mask
            const int jc = nt * 16 + (lane & 15);
            const int aj = jc >> 4, hj = (jc >> 2) & 3, wj = jc & 3;
            const int sel = (lane >> 4) * 8;
#pragma unroll
            for (int r = 0; r < 8; ++r) {
                int m  = mt * 16 + sel + r;
                int ai = m >> 4, hi_ = (m >> 2) & 3, wi_ = m & 3;
                int rpi = (ai - aj + 4) * 49 + (hi_ - hj + 3) * 7 + (wi_ - wj + 3);
                float s = acc[r] + bias_table[rpi * 4 + h];
                if (ai >= 4 || aj >= 4) s = NEG_MAX;
                Pf[m * PS + jc] = s;
            }
        }
        __syncthreads();

        // row softmax over 80 cols; write bf16 probs padded to 96 cols
        for (int m = wv; m < 80; m += 8) {
            float v0 = Pf[m * PS + lane];
            float v1 = Pf[m * PS + lane + 32];
            float v2 = (lane + 64 < 80) ? Pf[m * PS + lane + 64] : NEG_MAX;
            float mx = fmaxf(v0, fmaxf(v1, v2));
#pragma unroll
            for (int off = 16; off > 0; off >>= 1)
                mx = fmaxf(mx, __shfl_xor(mx, off, 32));
            float e0 = __expf(v0 - mx);
            float e1 = __expf(v1 - mx);
            float e2 = (lane + 64 < 80) ? __expf(v2 - mx) : 0.0f;
            float sm = e0 + e1 + e2;
#pragma unroll
            for (int off = 16; off > 0; off >>= 1)
                sm += __shfl_xor(sm, off, 32);
            float inv = 1.0f / sm;
            Pb[m * PS + lane]      = (bf16_t)(e0 * inv);
            Pb[m * PS + lane + 32] = (bf16_t)(e1 * inv);
            Pb[m * PS + lane + 64] = (bf16_t)((lane + 64 < 80) ? e2 * inv : 0.0f);
        }
        __syncthreads();

        // O_h = P V_h : 5x4 = 20 tiles, K = 96 (3 steps, zero padded)
        for (int tt = wv; tt < 20; tt += 8) {
            const int mt = tt / 4, nt = tt % 4;
            f32x8 acc = zero8();
            for (int ks = 0; ks < 3; ++ks) {
                bf16x16 afr = load_a_frag(Pb + mt * 16 * PS, PS, lane, ks * 32);
                bf16x16 bfr = load_b_frag(Vt + (h * 64 + nt * 16) * PS, PS, lane, ks * 32);
                acc = wmma_bf16(afr, bfr, acc);
            }
            const int d   = nt * 16 + (lane & 15);
            const int sel = (lane >> 4) * 8;
#pragma unroll
            for (int r = 0; r < 8; ++r) {
                int m = mt * 16 + sel + r;
                Ob[m * XS + h * 64 + d] = (bf16_t)acc[r];
            }
        }
        __syncthreads();
    }

    // ---- output projection: only 4 real agents (m < 64) -----------------
    // 4 m-tiles x 16 n-tiles = 64 tiles over 8 waves; K = 256 (8 steps)
    for (int tt = wv; tt < 64; tt += 8) {
        const int mt = tt >> 4;          // 0..3
        const int nt = tt & 15;          // 0..15
        f32x8 acc = zero8();
        for (int ks = 0; ks < 8; ++ks) {
            bf16x16 afr = load_a_frag(Ob + mt * 16 * XS, XS, lane, ks * 32);
            bf16x16 bfr = load_b_frag(woutT + nt * 16 * 256, 256, lane, ks * 32);
            acc = wmma_bf16(afr, bfr, acc);
        }
        const int c   = nt * 16 + (lane & 15);
        const int sel = (lane >> 4) * 8;
#pragma unroll
        for (int r = 0; r < 8; ++r) {
            int m  = mt * 16 + sel + r;
            int a  = m >> 4;
            int ph = (m >> 2) & 3;
            int pw = m & 3;
            out[(((a * 256 + c) * 128) + wh * 4 + ph) * 128 + ww * 4 + pw] = acc[r];
        }
    }
}

// ---------------------------------------------------------------------------
extern "C" void kernel_launch(void* const* d_in, const int* in_sizes, int n_in,
                              void* d_out, int out_size, void* d_ws, size_t ws_size,
                              hipStream_t stream) {
    const float* x     = (const float*)d_in[0];  // (4, 256, 128, 128)
    const float* w_qkv = (const float*)d_in[1];  // (256, 768)
    const float* w_out = (const float*)d_in[2];  // (256, 256)
    const float* bias  = (const float*)d_in[3];  // (441, 4)
    float* outp        = (float*)d_out;

    bf16_t* wqkvT = (bf16_t*)d_ws;                 // (768, 256) bf16
    bf16_t* woutT = wqkvT + 768 * 256;             // (256, 256) bf16

    // weight transpose + bf16 conversion (tiny, once per launch)
    transpose_to_bf16_kernel<<<(768 * 256 + 255) / 256, 256, 0, stream>>>(
        w_qkv, wqkvT, 256, 768);
    transpose_to_bf16_kernel<<<(256 * 256 + 255) / 256, 256, 0, stream>>>(
        w_out, woutT, 256, 256);

    // allow >64KB dynamic LDS (gfx1250 WGP has 320KB)
    (void)hipFuncSetAttribute((const void*)attn_win_kernel,
                              hipFuncAttributeMaxDynamicSharedMemorySize,
                              SMEM_BYTES);

    attn_win_kernel<<<1024, 256, SMEM_BYTES, stream>>>(x, wqkvT, woutT, bias, outp);
}
